// MHA_43765716746247
// MI455X (gfx1250) — matrix operand order repaired and checked by reading, not compiled, not run on hardware
//
#include <hip/hip_runtime.h>
#include <math.h>

typedef __attribute__((ext_vector_type(16))) _Float16 v16h;
typedef __attribute__((ext_vector_type(8)))  _Float16 v8h;
typedef __attribute__((ext_vector_type(8)))  float    v8f;

#define WMMA_F16(a, b, c) \
  __builtin_amdgcn_wmma_f32_16x16x32_f16(false, (a), false, (b), (short)0, (c), false, false)

// Problem constants (reference: B=2, S=2048, D=2048, H=32, hd=64, rotary=32)
constexpr int Bb   = 2;
constexpr int Ss   = 2048;
constexpr int Dd   = 2048;
constexpr int Hh   = 32;
constexpr int HD   = 64;
constexpr int Mrow = Bb * Ss;     // 4096
constexpr int NQKV = 3 * Dd;      // 6144
constexpr float LN1E4_D16 = 0.5756462732485115f; // ln(10000)/16

// ---------------------------------------------------------------------------
// A-fragment load: 16x32 f16 A matrix, row-major source, row stride in halfs.
// Layout (matches ldB): lanes 0-15 (M=lane): elems 0-15 = K0..15
//               lanes 16-31 (M=lane-16): elems 0-15 = K16..31
// `rowk` must point at &src[row*ld + kc].
// ---------------------------------------------------------------------------
__device__ __forceinline__ v16h ldA(const _Float16* rowk, int lane) {
  return *(const v16h*)(rowk + 16 * (lane >> 4));
}

// ---------------------------------------------------------------------------
// B-fragment load: 32x16 f16 B matrix from an N-major (transposed) source:
// lane supplies column n, elements are 16 contiguous K values.
// lanes 0-15: K = kc+0..15 ; lanes 16-31: K = kc+16..31.
// `colk` must point at &srcT[n*ld + kc]; ldB adds 16*(lane>>4).
// ---------------------------------------------------------------------------
__device__ __forceinline__ v16h ldB(const _Float16* colk, int lane) {
  return *(const v16h*)(colk + 16 * (lane >> 4));
}

// ===========================================================================
// Conversion kernels (one-time f32 -> f16 pack / transpose)
// ===========================================================================
__global__ void cvt_f32_to_f16(const float* __restrict__ in,
                               _Float16* __restrict__ out, int n) {
  int i = blockIdx.x * 256 + threadIdx.x;
  if (i < n) out[i] = (_Float16)in[i];
}

// outT[n*rows + k] = in[k*cols + n]   (in is [rows=K, cols=N] row-major)
__global__ void cvt_transpose_f16(const float* __restrict__ in,
                                  _Float16* __restrict__ outT,
                                  int rows, int cols) {
  int i = blockIdx.x * 256 + threadIdx.x;
  if (i < rows * cols) {
    int n = i / rows;
    int k = i - n * rows;
    outT[i] = (_Float16)in[k * cols + n];
  }
}

// ===========================================================================
// QKV GEMM + bias + RoPE.  Each wave: 16 rows x 64 cols (one head of q/k/v).
// grid.x = 256 M-tiles * 12 column-blocks, 256 threads (8 waves).
// Writes: Q,K as f16 [B,H,S,hd] (K pre-scaled by 1/8), V as f16 [B,H,hd,S].
// ===========================================================================
__global__ __launch_bounds__(256) void qkv_rope_kernel(
    const _Float16* __restrict__ x16, const _Float16* __restrict__ WqkvT,
    const float* __restrict__ bqkv, _Float16* __restrict__ Qb,
    _Float16* __restrict__ Kb, _Float16* __restrict__ Vb) {
  const int lane = threadIdx.x & 31;
  const int wave = threadIdx.x >> 5;
  const int g    = lane >> 4;
  const int nn   = lane & 15;
  const int mtile = blockIdx.x & 255;   // 0..255
  const int cblk  = blockIdx.x >> 8;    // 0..11
  const int cg    = cblk * 8 + wave;    // 0..95  (column group of 64)
  const int m0    = mtile * 16;
  const int n0    = cg * 64;

  v8f acc[4] = {};
  const _Float16* arow = x16 + (size_t)(m0 + nn) * Dd;
#pragma unroll 4
  for (int kc = 0; kc < Dd; kc += 32) {
    v16h a = ldA(arow + kc, lane);
#pragma unroll
    for (int t = 0; t < 4; ++t) {
      v16h b = ldB(WqkvT + (size_t)(n0 + t * 16 + nn) * Dd + kc, lane);
      acc[t] = WMMA_F16(a, b, acc[t]);
    }
  }

  // bias
#pragma unroll
  for (int t = 0; t < 4; ++t) {
    float bb = bqkv[n0 + t * 16 + nn];
#pragma unroll
    for (int v = 0; v < 8; ++v) acc[t][v] += bb;
  }

  const int type = cg >> 5;       // 0=q, 1=k, 2=v
  const int h    = cg & 31;
  const int bi   = m0 / Ss;
  const int s0   = m0 % Ss;

  if (type < 2) {
    // RoPE on dims [0,32): pair (d, d+16) lives in acc0/acc1 at same slot.
    const float inv = __expf(-(float)nn * LN1E4_D16);
#pragma unroll
    for (int v = 0; v < 8; ++v) {
      float sp = (float)(s0 + v + 8 * g);
      float sn, cs;
      __sincosf(sp * inv, &sn, &cs);
      float u1 = acc[0][v], u2 = acc[1][v];
      acc[0][v] = u1 * cs - u2 * sn;
      acc[1][v] = u1 * sn + u2 * cs;
    }
    const float sc = (type == 1) ? 0.125f : 1.0f;  // k pre-scaled by 1/sqrt(64)
    _Float16* dst = (type == 0) ? Qb : Kb;
#pragma unroll
    for (int t = 0; t < 4; ++t)
#pragma unroll
      for (int v = 0; v < 8; ++v) {
        int s = s0 + v + 8 * g;
        int d = t * 16 + nn;
        dst[(size_t)((bi * Hh + h) * Ss + s) * HD + d] =
            (_Float16)(acc[t][v] * sc);
      }
  } else {
    // V stored transposed: [B,H,hd,S]
#pragma unroll
    for (int t = 0; t < 4; ++t)
#pragma unroll
      for (int v = 0; v < 8; ++v) {
        int s = s0 + v + 8 * g;
        int d = t * 16 + nn;
        Vb[(size_t)((bi * Hh + h) * HD + d) * Ss + s] = (_Float16)acc[t][v];
      }
  }
}

// ===========================================================================
// Causal flash attention. One wave per 16-query tile; online softmax;
// P is re-laid-out C->A through a 1KB/wave LDS buffer.
// grid.x = (B*H*S/16)/8 = 1024, 256 threads.
// Output O16: f16 [B*S, D] with column = h*64 + d.
// ===========================================================================
__global__ __launch_bounds__(256) void flash_attn_kernel(
    const _Float16* __restrict__ Qb, const _Float16* __restrict__ Kb,
    const _Float16* __restrict__ Vb, _Float16* __restrict__ O16) {
  __shared__ _Float16 plds[8][16 * 32];

  const int lane = threadIdx.x & 31;
  const int wave = threadIdx.x >> 5;
  const int g    = lane >> 4;
  const int nn   = lane & 15;
  const int wg   = blockIdx.x * 8 + wave;
  const int qt   = wg & 127;       // S/16 = 128 query tiles
  const int bh   = wg >> 7;        // 0..63
  const int q0   = qt * 16;

  const _Float16* Qp = Qb + (size_t)bh * Ss * HD;
  const _Float16* Kp = Kb + (size_t)bh * Ss * HD;
  const _Float16* Vp = Vb + (size_t)bh * HD * Ss;

  // Q A-fragments for hd chunks [0,32) and [32,64)
  const v16h aq0 = ldA(Qp + (size_t)(q0 + nn) * HD + 0, lane);
  const v16h aq1 = ldA(Qp + (size_t)(q0 + nn) * HD + 32, lane);

  v8f o[4] = {};
  float mrun[8], lrun[8];
#pragma unroll
  for (int v = 0; v < 8; ++v) { mrun[v] = -__builtin_inff(); lrun[v] = 0.0f; }

  _Float16* pw = &plds[wave][0];
  const int nb = (q0 + 47) >> 5;   // 32-key blocks covering keys <= q0+15

  for (int jb = 0; jb < nb; ++jb) {
    const int j0 = jb * 32;

    // scores: two 16x16 tiles (keys j0..+15, j0+16..+31), K-pre-scaled
    v8f sc0 = {}, sc1 = {};
    {
      v16h b00 = ldB(Kp + (size_t)(j0 + nn) * HD + 0, lane);
      v16h b01 = ldB(Kp + (size_t)(j0 + nn) * HD + 32, lane);
      sc0 = WMMA_F16(aq0, b00, sc0);
      sc0 = WMMA_F16(aq1, b01, sc0);
      v16h b10 = ldB(Kp + (size_t)(j0 + 16 + nn) * HD + 0, lane);
      v16h b11 = ldB(Kp + (size_t)(j0 + 16 + nn) * HD + 32, lane);
      sc1 = WMMA_F16(aq0, b10, sc1);
      sc1 = WMMA_F16(aq1, b11, sc1);
    }

    // causal mask (matches reference's -10000 fill)
#pragma unroll
    for (int v = 0; v < 8; ++v) {
      const int qrow = q0 + v + 8 * g;
      if (j0 + nn > qrow)      sc0[v] = -10000.0f;
      if (j0 + 16 + nn > qrow) sc1[v] = -10000.0f;
    }

    // online softmax: row reductions across the 16-lane N-groups
    float corr[8];
#pragma unroll
    for (int v = 0; v < 8; ++v) {
      float mx = fmaxf(sc0[v], sc1[v]);
      mx = fmaxf(mx, __shfl_xor(mx, 1, 32));
      mx = fmaxf(mx, __shfl_xor(mx, 2, 32));
      mx = fmaxf(mx, __shfl_xor(mx, 4, 32));
      mx = fmaxf(mx, __shfl_xor(mx, 8, 32));
      const float mnew = fmaxf(mrun[v], mx);
      const float c = __expf(mrun[v] - mnew);
      mrun[v] = mnew;
      const float p0 = __expf(sc0[v] - mnew);
      const float p1 = __expf(sc1[v] - mnew);
      sc0[v] = p0; sc1[v] = p1;
      float rs = p0 + p1;
      rs += __shfl_xor(rs, 1, 32);
      rs += __shfl_xor(rs, 2, 32);
      rs += __shfl_xor(rs, 4, 32);
      rs += __shfl_xor(rs, 8, 32);
      lrun[v] = lrun[v] * c + rs;
      corr[v] = c;
    }
#pragma unroll
    for (int t = 0; t < 4; ++t)
#pragma unroll
      for (int v = 0; v < 8; ++v) o[t][v] *= corr[v];

    // P: C-layout (lane=col) -> LDS -> A-layout (lane=row)
#pragma unroll
    for (int v = 0; v < 8; ++v) {
      const int r = v + 8 * g;
      pw[r * 32 + nn]      = (_Float16)sc0[v];
      pw[r * 32 + 16 + nn] = (_Float16)sc1[v];
    }
    asm volatile("s_wait_dscnt 0" ::: "memory");   // wave-local LDS RAW fence
    v16h pa = ldA(pw + nn * 32, lane);

    // O += P x V  (V transposed: contiguous key loads)
#pragma unroll
    for (int t = 0; t < 4; ++t) {
      v16h vb = ldB(Vp + (size_t)(t * 16 + nn) * Ss + j0, lane);
      o[t] = WMMA_F16(pa, vb, o[t]);
    }
  }

  // normalize and store O16 [B*S, D], col = h*64 + d
  const int bi = bh >> 5, h = bh & 31;
#pragma unroll
  for (int t = 0; t < 4; ++t)
#pragma unroll
    for (int v = 0; v < 8; ++v) {
      const int qrow = q0 + v + 8 * g;
      const int d = t * 16 + nn;
      O16[(size_t)(bi * Ss + qrow) * Dd + h * HD + d] =
          (_Float16)(o[t][v] / lrun[v]);
    }
}

// ===========================================================================
// Output projection: out = O16 @ Wout + b_out, f32 result.
// grid.x = 256 M-tiles * 4 column-blocks, 256 threads (8 waves).
// ===========================================================================
__global__ __launch_bounds__(256) void out_proj_kernel(
    const _Float16* __restrict__ O16, const _Float16* __restrict__ WoutT,
    const float* __restrict__ bout, float* __restrict__ out) {
  const int lane = threadIdx.x & 31;
  const int wave = threadIdx.x >> 5;
  const int g    = lane >> 4;
  const int nn   = lane & 15;
  const int mtile = blockIdx.x & 255;
  const int cblk  = blockIdx.x >> 8;   // 0..3
  const int cg    = cblk * 8 + wave;   // 0..31
  const int m0    = mtile * 16;
  const int n0    = cg * 64;

  v8f acc[4] = {};
  const _Float16* arow = O16 + (size_t)(m0 + nn) * Dd;
#pragma unroll 4
  for (int kc = 0; kc < Dd; kc += 32) {
    v16h a = ldA(arow + kc, lane);
#pragma unroll
    for (int t = 0; t < 4; ++t) {
      v16h b = ldB(WoutT + (size_t)(n0 + t * 16 + nn) * Dd + kc, lane);
      acc[t] = WMMA_F16(a, b, acc[t]);
    }
  }
#pragma unroll
  for (int t = 0; t < 4; ++t) {
    const float bb = bout[n0 + t * 16 + nn];
#pragma unroll
    for (int v = 0; v < 8; ++v) {
      const int row = m0 + v + 8 * g;
      out[(size_t)row * Dd + n0 + t * 16 + nn] = acc[t][v] + bb;
    }
  }
}

// ===========================================================================
// Launch
// ===========================================================================
extern "C" void kernel_launch(void* const* d_in, const int* in_sizes, int n_in,
                              void* d_out, int out_size, void* d_ws,
                              size_t ws_size, hipStream_t stream) {
  (void)in_sizes; (void)n_in; (void)out_size; (void)ws_size;
  const float* x    = (const float*)d_in[0];
  const float* Wqkv = (const float*)d_in[1];
  const float* bqkv = (const float*)d_in[2];
  const float* Wout = (const float*)d_in[3];
  const float* bout = (const float*)d_in[4];
  float* out = (float*)d_out;

  char* ws = (char*)d_ws;
  _Float16* x16   = (_Float16*)ws;  ws += (size_t)Mrow * Dd * 2;     // 16 MB
  _Float16* WqkvT = (_Float16*)ws;  ws += (size_t)NQKV * Dd * 2;     // 24 MB
  _Float16* WoutT = (_Float16*)ws;  ws += (size_t)Dd * Dd * 2;       //  8 MB
  _Float16* Qb    = (_Float16*)ws;  ws += (size_t)Mrow * Dd * 2;     // 16 MB
  _Float16* Kb    = (_Float16*)ws;  ws += (size_t)Mrow * Dd * 2;     // 16 MB
  _Float16* Vb    = (_Float16*)ws;  ws += (size_t)Mrow * Dd * 2;     // 16 MB
  _Float16* O16   = (_Float16*)ws;  ws += (size_t)Mrow * Dd * 2;     // 16 MB

  {
    int n = Mrow * Dd;
    cvt_f32_to_f16<<<(n + 255) / 256, 256, 0, stream>>>(x, x16, n);
  }
  {
    int n = Dd * NQKV;
    cvt_transpose_f16<<<(n + 255) / 256, 256, 0, stream>>>(Wqkv, WqkvT, Dd, NQKV);
  }
  {
    int n = Dd * Dd;
    cvt_transpose_f16<<<(n + 255) / 256, 256, 0, stream>>>(Wout, WoutT, Dd, Dd);
  }

  qkv_rope_kernel<<<256 * 12, 256, 0, stream>>>(x16, WqkvT, bqkv, Qb, Kb, Vb);
  flash_attn_kernel<<<(Bb * Hh * (Ss / 16)) / 8, 256, 0, stream>>>(Qb, Kb, Vb, O16);
  out_proj_kernel<<<256 * 4, 256, 0, stream>>>(O16, WoutT, bout, out);
}
